// RowLSTM_72430328481588
// MI455X (gfx1250) — matrix operand order repaired and checked by reading, not compiled, not run on hardware
//
#include <hip/hip_runtime.h>
#include <hip/hip_bf16.h>

typedef __attribute__((ext_vector_type(16))) __bf16 v16bf;
typedef __attribute__((ext_vector_type(8)))  __bf16 v8bf;
typedef __attribute__((ext_vector_type(8)))  float  v8f;

#define Bn  16
#define Cc  64
#define Hh  64
#define Ww  64
#define HID 128
#define G4  512   // 4*HID
#define KWW 3

// Per-lane element index -> K mapping for the 16x32 bf16 A fragment
// (contiguous order: element i holds K = 16*half + i, matching the B fragments)
__device__ inline int amapK(int i, int half) {
  return 16 * half + i;
}

__device__ inline float sigm(float x) { return 1.0f / (1.0f + __expf(-x)); }

// ---------------------------------------------------------------------------
// Kernel 0: pack Wh / Wi into A-fragment bf16 layout; biasSum = bi + bh.
// Wh: 32 Mtiles x 4 Ktiles (K=HID). Wi: 32 Mtiles x 6 Ktiles (K = kw*64 + c).
// Fragment f stored as 32 lanes x 16 halves contiguous (32B per lane).
// ---------------------------------------------------------------------------
__global__ void pack_weights(const float* __restrict__ Wi, const float* __restrict__ bi,
                             const float* __restrict__ Wh, const float* __restrict__ bh,
                             __bf16* __restrict__ whPack, __bf16* __restrict__ wiPack,
                             float* __restrict__ biasSum) {
  int idx = blockIdx.x * blockDim.x + threadIdx.x;
  if (idx < 32 * 4 * 512) {   // Wh pack
    int f = idx >> 9, r = idx & 511;
    int lane = r >> 4, i = r & 15;
    int mt = f >> 2, kt = f & 3;
    int half = lane >> 4;
    int M = mt * 16 + (lane & 15);
    int K = kt * 32 + amapK(i, half);
    whPack[idx] = (__bf16)Wh[M * HID + K];
  }
  if (idx < 32 * 6 * 512) {   // Wi pack (K index = kw*64 + c)
    int f = idx >> 9, r = idx & 511;
    int lane = r >> 4, i = r & 15;
    int mt = f / 6, kt = f % 6;
    int half = lane >> 4;
    int M = mt * 16 + (lane & 15);
    int K = kt * 32 + amapK(i, half);
    int kw = K >> 6, c = K & 63;
    wiPack[idx] = (__bf16)Wi[(M * Cc + c) * KWW + kw];
  }
  if (idx < G4) biasSum[idx] = bi[idx] + bh[idx];
}

// ---------------------------------------------------------------------------
// Kernel 1: i2h conv as WMMA GEMM. One WG per (b,row). M=512, K=192, N=64.
// Output stored in C-fragment layout (bf16) for direct consumption by row_lstm:
//   i2hPack[b][wt][row][mtile][lane][8]
// ---------------------------------------------------------------------------
__global__ __launch_bounds__(256)
void conv_i2h(const float* __restrict__ x, const __bf16* __restrict__ wiPack,
              __bf16* __restrict__ i2hPack) {
  __shared__ __align__(32) __bf16 xs[66 * 64];  // [w_padded][c], bf16
  const int b   = blockIdx.x >> 6;
  const int row = blockIdx.x & 63;
  const int tid = threadIdx.x;

  // Stage padded x row (all channels) into LDS, f32 -> bf16.
  // x is streamed exactly once -> nontemporal loads keep it out of L2.
  for (int e = tid; e < 66 * 64; e += 256) {
    int j = e >> 6, c = e & 63;
    float v = 0.0f;
    if (j >= 1 && j <= 64)
      v = __builtin_nontemporal_load(
            &x[(((size_t)b * Cc + c) * Hh + row) * Ww + (j - 1)]);
    xs[e] = (__bf16)v;
  }
  __syncthreads();

  const int wave = tid >> 5, lane = tid & 31;
  const int n = lane & 15, half = lane >> 4;

  v8f acc[4][4];
  v8f zero;
  #pragma unroll
  for (int e = 0; e < 8; ++e) zero[e] = 0.0f;
  #pragma unroll
  for (int m = 0; m < 4; ++m)
    #pragma unroll
    for (int nt = 0; nt < 4; ++nt) acc[m][nt] = zero;

  #pragma unroll
  for (int kt = 0; kt < 6; ++kt) {
    v16bf a[4], bb[4];
    #pragma unroll
    for (int m = 0; m < 4; ++m) {
      int mt = wave * 4 + m;
      a[m] = *(const v16bf*)(wiPack + ((size_t)(mt * 6 + kt) * 512 + lane * 16));
    }
    // B fragment: element i -> K = kt*32 + 16*half + i (contiguous c, fixed kw)
    int baseK = kt * 32 + 16 * half;
    int kw = baseK >> 6, c0 = baseK & 63;
    #pragma unroll
    for (int nt = 0; nt < 4; ++nt) {
      int j = nt * 16 + n + kw;           // padded column = w + kw
      bb[nt] = *(const v16bf*)&xs[j * 64 + c0];
    }
    #pragma unroll
    for (int m = 0; m < 4; ++m)
      #pragma unroll
      for (int nt = 0; nt < 4; ++nt)
        acc[m][nt] = __builtin_amdgcn_wmma_f32_16x16x32_bf16(
            false, a[m], false, bb[nt], (short)0, acc[m][nt], false, false);
  }

  #pragma unroll
  for (int m = 0; m < 4; ++m) {
    int mt = wave * 4 + m;
    #pragma unroll
    for (int nt = 0; nt < 4; ++nt) {
      v8bf ob;
      #pragma unroll
      for (int e = 0; e < 8; ++e) ob[e] = (__bf16)acc[m][nt][e];
      size_t off = (((((size_t)b * 4 + nt) * Hh + row) * 32 + mt) * 32 + lane) * 8;
      *(v8bf*)(i2hPack + off) = ob;   // regular store: consumed soon via L2
    }
  }
}

// ---------------------------------------------------------------------------
// Kernel 2: row LSTM scan. One WG per (b, wt): 16 independent (b,w) columns.
// Per step: gates = i2h + bias + Wh @ h via 16 WMMAs/wave; wave v owns
// mtiles {v, 8+v, 16+v, 24+v} => all 4 gates of hid slice [16v,16v+16) in-wave.
// i2h fragments are software-pipelined one row ahead (+ prefetch two ahead);
// h round-trips through LDS laid out [col][hid] for contiguous stores/gathers.
// ---------------------------------------------------------------------------
__global__ __launch_bounds__(256)
void row_lstm(const __bf16* __restrict__ whPack, const float* __restrict__ biasSum,
              const __bf16* __restrict__ i2hPack, float* __restrict__ out) {
  __shared__ __align__(32) __bf16 hlds[16 * HID];  // [col][hid]
  const int b  = blockIdx.x >> 2;
  const int wt = blockIdx.x & 3;
  const int tid = threadIdx.x;
  const int wave = tid >> 5, lane = tid & 31;
  const int n = lane & 15, half = lane >> 4;

  for (int e = tid; e < 16 * HID; e += 256) hlds[e] = (__bf16)0.0f;

  // Resident Wh A-fragments + bias for this wave's 4 gate tiles
  v16bf whA[4][4];
  float bias[4][8];
  #pragma unroll
  for (int g = 0; g < 4; ++g) {
    int mt = g * 8 + wave;
    #pragma unroll
    for (int kt = 0; kt < 4; ++kt)
      whA[g][kt] = *(const v16bf*)(whPack + ((size_t)(mt * 4 + kt) * 512 + lane * 16));
    #pragma unroll
    for (int r = 0; r < 8; ++r)
      bias[g][r] = biasSum[mt * 16 + half * 8 + r];
  }

  // Per-gate i2h fragment base pointers for this lane (row stride = 32*32*8 halves)
  const __bf16* i2hBase[4];
  #pragma unroll
  for (int g = 0; g < 4; ++g) {
    int mt = g * 8 + wave;
    i2hBase[g] = i2hPack +
        ((((size_t)b * 4 + wt) * Hh * 32 + mt) * 32 + lane) * 8;
  }
  const size_t rowStride = (size_t)32 * 32 * 8;  // halves per row

  float cstate[8];
  #pragma unroll
  for (int r = 0; r < 8; ++r) cstate[r] = 0.0f;

  // Preload row 0 i2h fragments
  v8bf ivcur[4];
  #pragma unroll
  for (int g = 0; g < 4; ++g)
    ivcur[g] = *(const v8bf*)(i2hBase[g]);

  __syncthreads();

  for (int row = 0; row < Hh; ++row) {
    // Gather previous h as B fragments: element i -> K = kt*32 + 16*half + i
    v16bf hB[4];
    #pragma unroll
    for (int kt = 0; kt < 4; ++kt)
      hB[kt] = *(const v16bf*)&hlds[n * HID + kt * 32 + 16 * half];
    __syncthreads();  // all reads of hlds done before anyone overwrites

    // acc = bias + i2h (current row, preloaded last iteration)
    v8f acc[4];
    #pragma unroll
    for (int g = 0; g < 4; ++g) {
      #pragma unroll
      for (int r = 0; r < 8; ++r) acc[g][r] = bias[g][r] + (float)ivcur[g][r];
    }

    // Issue next row's i2h loads + prefetch row+2 so they overlap the WMMAs
    v8bf ivnext[4];
    if (row + 1 < Hh) {
      #pragma unroll
      for (int g = 0; g < 4; ++g)
        ivnext[g] = *(const v8bf*)(i2hBase[g] + (size_t)(row + 1) * rowStride);
    }
    if (row + 2 < Hh) {
      #pragma unroll
      for (int g = 0; g < 4; ++g)
        __builtin_prefetch(i2hBase[g] + (size_t)(row + 2) * rowStride, 0, 0);
    }

    #pragma unroll
    for (int g = 0; g < 4; ++g)
      #pragma unroll
      for (int kt = 0; kt < 4; ++kt)
        acc[g] = __builtin_amdgcn_wmma_f32_16x16x32_bf16(
            false, whA[g][kt], false, hB[kt], (short)0, acc[g], false, false);

    // LSTM cell (f32): acc[0]=i, [1]=f, [2]=o, [3]=g
    float hval[8];
    #pragma unroll
    for (int r = 0; r < 8; ++r) {
      float ig = sigm(acc[0][r]);
      float fg = sigm(acc[1][r]);
      float og = sigm(acc[2][r]);
      float gg = tanhf(acc[3][r]);
      float c2 = fg * cstate[r] + ig * gg;
      cstate[r] = c2;
      hval[r] = og * tanhf(c2);
    }

    // Write h for next step: [col][hid] => 8 contiguous halves per lane
    v8bf hb;
    #pragma unroll
    for (int r = 0; r < 8; ++r) hb[r] = (__bf16)hval[r];
    *(v8bf*)&hlds[n * HID + wave * 16 + half * 8] = hb;

    // Emit output h (B, HID, H, W), f32; write-once -> nontemporal
    #pragma unroll
    for (int r = 0; r < 8; ++r) {
      int hid = wave * 16 + half * 8 + r;
      __builtin_nontemporal_store(
          hval[r], &out[(((size_t)b * HID + hid) * Hh + row) * Ww + wt * 16 + n]);
    }

    #pragma unroll
    for (int g = 0; g < 4; ++g) ivcur[g] = ivnext[g];

    __syncthreads();  // new h visible before next gather
  }
}

// ---------------------------------------------------------------------------
extern "C" void kernel_launch(void* const* d_in, const int* in_sizes, int n_in,
                              void* d_out, int out_size, void* d_ws, size_t ws_size,
                              hipStream_t stream) {
  const float* x  = (const float*)d_in[0];
  const float* Wi = (const float*)d_in[1];
  const float* bi = (const float*)d_in[2];
  const float* Wh = (const float*)d_in[3];
  const float* bh = (const float*)d_in[4];
  float* out = (float*)d_out;

  char* ws = (char*)d_ws;
  __bf16* whPack  = (__bf16*)(ws);                          // 128 KiB
  __bf16* wiPack  = (__bf16*)(ws + 131072);                 // 192 KiB
  float*  biasSum = (float*)(ws + 131072 + 196608);         // 2 KiB
  __bf16* i2hPack = (__bf16*)(ws + 131072 + 196608 + 2048); // 64 MiB

  pack_weights<<<(32 * 6 * 512 + 255) / 256, 256, 0, stream>>>(
      Wi, bi, Wh, bh, whPack, wiPack, biasSum);
  conv_i2h<<<Bn * Hh, 256, 0, stream>>>(x, wiPack, i2hPack);
  row_lstm<<<Bn * (Ww / 16), 256, 0, stream>>>(whPack, biasSum, i2hPack, out);
}
